// ComparingLoss_34548716929310
// MI455X (gfx1250) — compile-verified, matrix operand order checked by reading, not run on hardware
//
#include <hip/hip_runtime.h>

// ---------------------------------------------------------------------------
// ComparingLoss on MI455X (gfx1250, wave32)
//
//   s = sum_{i<j} disc(i,j) * (d_i + d_j)   with  d = |x - label|,
//   disc(i,j) = (x_i > x_j) XOR (label_i > label_j)
//
// For tie-free inputs disc is symmetric, so  s = sum_i d_i * n_i  where
// n_i = #{ j != i : disc(i,j) }.  n_i is a row-sum of a boolean T x T matrix,
// computed 16x64 pairs at a time with V_WMMA_I32_16X16X64_IU8 (B = ones:
// D = A x 1 = row sums; byte position within a row is then irrelevant).
// ---------------------------------------------------------------------------

typedef int v8i __attribute__((ext_vector_type(8)));

#define T_ELEMS 8192
#define J_STRIP 2048        // j-range staged in LDS per block (16 KB)
#define BLOCK_THREADS 256   // 8 waves (wave32)
#define WAVES_PER_BLOCK 8

__global__ __launch_bounds__(BLOCK_THREADS)
void pair_disc_count_kernel(const float* __restrict__ x,
                            const float* __restrict__ lab,
                            float* __restrict__ ws_sum) {
    // Interleaved {x_k, l_k, x_{k+1}, l_{k+1}} so one ds_load_b128 = 2 pairs.
    __shared__ float4 sTile[J_STRIP / 2];   // 16 KB of 320 KB/WGP

    const int tid  = threadIdx.x;
    const int lane = tid & 31;
    const int wave = tid >> 5;
    const int m    = lane & 15;     // WMMA A row = lane & 15 (ISA 8-bit A layout)
    const int h    = lane >> 4;     // half-wave: disjoint j sub-ranges per half

    const int js = blockIdx.y * J_STRIP;
    const int i0 = (blockIdx.x * WAVES_PER_BLOCK + wave) * 16;

    // ---- stage the j-strip into LDS (256 threads x 4 float4 = 1024 float4) ----
    #pragma unroll
    for (int c = 0; c < 4; ++c) {
        const int q = tid * 4 + c;          // float4 index 0..1023
        const int k = js + 2 * q;
        sTile[q] = make_float4(x[k], lab[k], x[k + 1], lab[k + 1]);
    }

    // ---- per-lane row operands ----
    const int   i  = i0 + m;
    const float xi = x[i];
    const float li = lab[i];

    // d-values for this half's 8 rows (rows v + 8h live in this half's D VGPRs)
    float d8[8];
    #pragma unroll
    for (int v = 0; v < 8; ++v) {
        const int r = i0 + 8 * h + v;
        d8[v] = fabsf(x[r] - lab[r]);
    }

    __syncthreads();

    v8i acc = {0, 0, 0, 0, 0, 0, 0, 0};
    const v8i bones = {0x01010101, 0x01010101, 0x01010101, 0x01010101,
                       0x01010101, 0x01010101, 0x01010101, 0x01010101};

    // ---- main loop: 64 j per WMMA, rows = 16 i's of this wave ----
    for (int jc = 0; jc < J_STRIP; jc += 64) {
        // lane's j sub-range: jc + 32*h + [0,32)   (float4 index = pair/2)
        const float4* base = &sTile[(jc + 32 * h) >> 1];
        v8i a;
        #pragma unroll
        for (int v = 0; v < 8; ++v) {
            const float4 p01 = base[2 * v];       // pairs (4v+0, 4v+1)
            const float4 p23 = base[2 * v + 1];   // pairs (4v+2, 4v+3)
            const unsigned c0 = ((xi > p01.x) != (li > p01.y)) ? 1u : 0u;
            const unsigned c1 = ((xi > p01.z) != (li > p01.w)) ? 1u : 0u;
            const unsigned c2 = ((xi > p23.x) != (li > p23.y)) ? 1u : 0u;
            const unsigned c3 = ((xi > p23.z) != (li > p23.w)) ? 1u : 0u;
            a[v] = (int)(c0 | (c1 << 8) | (c2 << 16) | (c3 << 24));
        }
        // D[m][*] += row sums of the 16x64 0/1 tile
        acc = __builtin_amdgcn_wmma_i32_16x16x64_iu8(
            /*sgn_a=*/false, a, /*sgn_b=*/false, bones, acc,
            /*reuse_a=*/false, /*reuse_b=*/false);
    }

    // ---- epilogue: partial = sum_m d_m * cnt_m for this wave's 16 rows ----
    // D layout: lane holds rows v + 8h in VGPR v, replicated over 16 columns,
    // so the wave-sum below overcounts by exactly 16x.
    float tmp = 0.0f;
    #pragma unroll
    for (int v = 0; v < 8; ++v) tmp += d8[v] * (float)acc[v];

    #pragma unroll
    for (int off = 16; off > 0; off >>= 1)
        tmp += __shfl_down(tmp, off, 32);

    if (lane == 0)
        atomicAdd(ws_sum, tmp * (1.0f / 16.0f));
}

__global__ void finalize_kernel(const float* __restrict__ ws_sum,
                                float* __restrict__ out) {
    out[0] = ws_sum[0] * (1.0f / (float)T_ELEMS);
}

extern "C" void kernel_launch(void* const* d_in, const int* in_sizes, int n_in,
                              void* d_out, int out_size, void* d_ws, size_t ws_size,
                              hipStream_t stream) {
    const float* x   = (const float*)d_in[0];
    const float* lab = (const float*)d_in[1];
    float*       out = (float*)d_out;
    float*       ws  = (float*)d_ws;

    // Scratch accumulator must start at zero every call (graph-capture safe).
    hipMemsetAsync(ws, 0, sizeof(float), stream);

    dim3 grid(T_ELEMS / (16 * WAVES_PER_BLOCK),   // 64 row-block groups
              T_ELEMS / J_STRIP);                 // 4 j-strips
    pair_disc_count_kernel<<<grid, BLOCK_THREADS, 0, stream>>>(x, lab, ws);
    finalize_kernel<<<1, 1, 0, stream>>>(ws, out);
}
